// MiddleBlock_49143015801181
// MI455X (gfx1250) — matrix operand order repaired and checked by reading, not compiled, not run on hardware
//
#include <hip/hip_runtime.h>
#include <hip/hip_bf16.h>

#define B_   8
#define C_   512
#define N_   1024      // 32*32 spatial
#define COND_ 1280
#define HEADS_ 8
#define D_   64
#define GROUPS_ 64     // 512/8 -> 8 channels per group

typedef _Float16 v16h __attribute__((ext_vector_type(16)));
typedef _Float16 v8h  __attribute__((ext_vector_type(8)));
typedef _Float16 v4h  __attribute__((ext_vector_type(4)));
typedef float    v8f  __attribute__((ext_vector_type(8)));
typedef float    v4f  __attribute__((ext_vector_type(4)));

// ---- force global (AS1) memory ops so bulk traffic never lowers to FLAT ----
__device__ __forceinline__ v4f ldg_v4f(const float* p) {
    return *(const v4f __attribute__((address_space(1))) *)p;
}
__device__ __forceinline__ float ldg_f(const float* p) {
    return *(const float __attribute__((address_space(1))) *)p;
}
__device__ __forceinline__ void stg_f(float* p, float v) {
    *(float __attribute__((address_space(1))) *)p = v;
}
__device__ __forceinline__ void stg_v4f(float* p, v4f v) {
    *(v4f __attribute__((address_space(1))) *)p = v;
}

union Frag16 { v16h v; v8h h[2]; };

__device__ __forceinline__ v8f wmma_f16f32(const Frag16& a, const Frag16& b, v8f c) {
    // D = A(16x32 f16) * B(32x16 f16) + C(16x16 f32)
    return __builtin_amdgcn_wmma_f32_16x16x32_f16(
        false, a.v, false, b.v, (short)0, c, false, false);
}

// ---------------------------------------------------------------------------
// WMMA GEMM:  Out[b][m][n] = sum_k W[m][k] * X(b,k,n) + bias[m] (+addend)
// IM2COL=0: X is [B][K][N] fp32; IM2COL=1: 3x3 conv, K=512*9, X=[B][512][32][32]
// Block 256 threads (8 waves), tile 128x128, K-step 32.
// Software pipeline: regs prefetch next tile while WMMAs run; LDS double buffer.
// ---------------------------------------------------------------------------
#define TS_K 32
#define ASTR 40   // half stride (80B, 16B-aligned fragment loads)
#define BSTR 40

template<int IM2COL, bool HASADD>
__global__ __launch_bounds__(256) void gemm_wmma(
    const float* __restrict__ W, const float* __restrict__ X,
    const float* __restrict__ bias, const float* __restrict__ addend,
    float* __restrict__ Out, int M, int K)
{
    __shared__ alignas(16) _Float16 As[2][128 * ASTR];
    __shared__ alignas(16) _Float16 Bs[2][128 * BSTR];

    const int t  = threadIdx.x;
    const int n0 = blockIdx.x * 128;
    const int m0 = blockIdx.y * 128;
    const int b  = blockIdx.z;

    X   += (size_t)b * C_ * N_;
    Out += (size_t)b * M * N_;
    const float* add = addend + (size_t)b * M * N_;

    const int wave = t >> 5, lane = t & 31;
    const int g = lane >> 4, lr = lane & 15;
    const int wm = (wave & 1) * 64;     // wave M offset in tile
    const int wn = (wave >> 1) * 32;    // wave N offset in tile

    // ---- hoisted A staging pointers: 4 float4 per thread ----
    const float* pA[4];
    int aofs[4];
    #pragma unroll
    for (int e = 0; e < 4; ++e) {
        int lin4 = t + e * 256;               // 0..1023 float4 slots
        int m  = lin4 >> 3;                   // 8 float4 per 32-wide row
        int kv = (lin4 & 7) * 4;
        pA[e]   = W + (size_t)(m0 + m) * K + kv;
        aofs[e] = m * ASTR + kv;
    }
    // ---- hoisted plain-mode B staging pointers ----
    const float* pB[4];
    int bofs[4];
    #pragma unroll
    for (int e = 0; e < 4; ++e) {
        int lin4 = t + e * 256;
        int kl  = lin4 >> 5;                  // 32 float4 per 128-wide row
        int nl4 = (lin4 & 31) * 4;
        pB[e]   = X + (size_t)kl * N_ + n0 + nl4;
        bofs[e] = nl4 * BSTR + kl;
    }

    v4f  aReg[4];
    v4f  bReg[4];     // plain mode
    float bIm[16];    // im2col mode

    auto loadRegs = [&](int k0) {
        #pragma unroll
        for (int e = 0; e < 4; ++e) { aReg[e] = ldg_v4f(pA[e]); pA[e] += TS_K; }
        __builtin_prefetch(pA[0], 0, 1);
        if constexpr (IM2COL) {
            #pragma unroll
            for (int e = 0; e < 16; ++e) {
                int lin = t + e * 256;
                int kl = lin >> 7;            // 0..31
                int nl = lin & 127;           // 0..127
                int kk = k0 + kl;
                int n  = n0 + nl;
                int cin = kk / 9, r = kk % 9;
                int dy = r / 3 - 1, dx = r % 3 - 1;
                int y = (n >> 5) + dy, x = (n & 31) + dx;
                bIm[e] = (y >= 0 && y < 32 && x >= 0 && x < 32)
                             ? ldg_f(&X[cin * N_ + y * 32 + x]) : 0.0f;
            }
        } else {
            #pragma unroll
            for (int e = 0; e < 4; ++e) {
                bReg[e] = ldg_v4f(pB[e]);
                pB[e] += (size_t)TS_K * N_;
            }
            __builtin_prefetch(pB[0], 0, 1);
        }
    };

    _Float16* ldsA = As[0]; _Float16* ldsA1 = As[1];
    _Float16* ldsB = Bs[0]; _Float16* ldsB1 = Bs[1];

    auto storeLds = [&]() {
        #pragma unroll
        for (int e = 0; e < 4; ++e)
            *(v4h*)&ldsA[aofs[e]] = __builtin_convertvector(aReg[e], v4h);
        if constexpr (IM2COL) {
            #pragma unroll
            for (int e = 0; e < 16; ++e) {
                int lin = t + e * 256;
                int kl = lin >> 7;
                int nl = lin & 127;
                ldsB[nl * BSTR + kl] = (_Float16)bIm[e];
            }
        } else {
            #pragma unroll
            for (int e = 0; e < 4; ++e) {
                v4h h = __builtin_convertvector(bReg[e], v4h);
                int base = bofs[e];
                ldsB[base]            = h.x;
                ldsB[base + BSTR]     = h.y;
                ldsB[base + 2 * BSTR] = h.z;
                ldsB[base + 3 * BSTR] = h.w;
            }
        }
    };

    v8f acc[4][2] = {};
    const int aFragOfs = (wm + lr) * ASTR + g * 8;
    const int bFragOfs = (wn + lr) * BSTR + g * 8;

    loadRegs(0);
    for (int k0 = 0; k0 < K; k0 += TS_K) {
        storeLds();                      // waits loadcnt, writes LDS[cur]
        __syncthreads();
        if (k0 + TS_K < K) loadRegs(k0 + TS_K);   // loads fly during WMMAs

        const _Float16* aF = ldsA + aFragOfs;
        const _Float16* bF = ldsB + bFragOfs;
        Frag16 bf[2];
        #pragma unroll
        for (int ni = 0; ni < 2; ++ni) {
            const _Float16* bp = bF + ni * 16 * BSTR;
            bf[ni].h[0] = *(const v8h*)(bp);
            bf[ni].h[1] = *(const v8h*)(bp + 16);
        }
        #pragma unroll
        for (int mi = 0; mi < 4; ++mi) {
            Frag16 af;
            const _Float16* ap = aF + mi * 16 * ASTR;
            af.h[0] = *(const v8h*)(ap);
            af.h[1] = *(const v8h*)(ap + 16);
            #pragma unroll
            for (int ni = 0; ni < 2; ++ni)
                acc[mi][ni] = wmma_f16f32(af, bf[ni], acc[mi][ni]);
        }
        // swap double buffers (pointer swap, no re-derivation)
        _Float16* tA = ldsA; ldsA = ldsA1; ldsA1 = tA;
        _Float16* tB = ldsB; ldsB = ldsB1; ldsB1 = tB;
    }

    // ---- epilogue: C/D layout -> VGPR r: M = r + 8*g, N = lane%16 ----
    #pragma unroll
    for (int mi = 0; mi < 4; ++mi)
        #pragma unroll
        for (int ni = 0; ni < 2; ++ni) {
            int m_base = m0 + wm + mi * 16 + 8 * g;
            int n = n0 + wn + ni * 16 + lr;
            #pragma unroll
            for (int r = 0; r < 8; ++r) {
                int m = m_base + r;
                float vv = acc[mi][ni][r] + ldg_f(&bias[m]);
                if constexpr (HASADD) vv += ldg_f(&add[(size_t)m * N_ + n]);
                stg_f(&Out[(size_t)m * N_ + n], vv);
            }
        }
}

// ---------------------------------------------------------------------------
// Conditioning projection: cond[b][c] = conditioning[b] . cpw[c] + cpb[c]
// ---------------------------------------------------------------------------
__global__ __launch_bounds__(256) void cond_proj_kernel(
    const float* __restrict__ cnd, const float* __restrict__ w,
    const float* __restrict__ bias, float* __restrict__ out)
{
    int idx = blockIdx.x * 256 + threadIdx.x;  // B_*C_ = 4096
    int b = idx >> 9, c = idx & 511;
    const float* cp = cnd + (size_t)b * COND_;
    const float* wp = w + (size_t)c * COND_;
    float s = ldg_f(&bias[c]);
    for (int j = 0; j < COND_; j += 4) {
        v4f a = ldg_v4f(cp + j), bb = ldg_v4f(wp + j);
        s += a.x * bb.x + a.y * bb.y + a.z * bb.z + a.w * bb.w;
    }
    stg_f(&out[idx], s);
}

// ---------------------------------------------------------------------------
// GroupNorm stats via wave32 shuffle + one LDS hop (2 barriers total)
// ---------------------------------------------------------------------------
__device__ __forceinline__ void gn_stats(float s1, float s2,
                                         float& mean, float& rstd) {
    #pragma unroll
    for (int o = 16; o > 0; o >>= 1) {
        s1 += __shfl_xor(s1, o, 32);
        s2 += __shfl_xor(s2, o, 32);
    }
    __shared__ float w1[8], w2[8];
    const int wave = threadIdx.x >> 5, lane = threadIdx.x & 31;
    if (lane == 0) { w1[wave] = s1; w2[wave] = s2; }
    __syncthreads();
    float t1 = 0.f, t2 = 0.f;
    #pragma unroll
    for (int i = 0; i < 8; ++i) { t1 += w1[i]; t2 += w2[i]; }
    mean = t1 * (1.f / 8192.f);
    float var = t2 * (1.f / 8192.f) - mean * mean;
    rstd = rsqrtf(var + 1e-5f);
    __syncthreads();
}

// GroupNorm (+SiLU).  One block per (batch, group); 8 ch x 1024 = 8192 elems.
// X tile cached in registers: read global once, write once.
__global__ __launch_bounds__(256) void gn_silu_kernel(
    const float* __restrict__ X, const float* __restrict__ gamma,
    const float* __restrict__ beta, float* __restrict__ Y)
{
    const int bg = blockIdx.x;
    const int b = bg >> 6, grp = bg & 63;
    const size_t base = ((size_t)b * C_ + grp * 8) * N_;
    const int tid = threadIdx.x;
    v4f xc[8];
    float s1 = 0.f, s2 = 0.f;
    #pragma unroll
    for (int e = 0; e < 8; ++e) {
        int i = tid + e * 256;
        v4f v = ldg_v4f(X + base + i * 4);
        xc[e] = v;
        s1 += v.x + v.y + v.z + v.w;
        s2 += v.x * v.x + v.y * v.y + v.z * v.z + v.w * v.w;
    }
    float mean, rstd;
    gn_stats(s1, s2, mean, rstd);
    #pragma unroll
    for (int e = 0; e < 8; ++e) {
        int i = tid + e * 256;
        int c = grp * 8 + (i >> 8);
        float gm = ldg_f(&gamma[c]), bt = ldg_f(&beta[c]);
        float ga = gm * rstd, be = bt - mean * rstd * gm;
        v4f v = xc[e], y;
        y.x = v.x * ga + be; y.y = v.y * ga + be;
        y.z = v.z * ga + be; y.w = v.w * ga + be;
        y.x = y.x / (1.f + __expf(-y.x));
        y.y = y.y / (1.f + __expf(-y.y));
        y.z = y.z / (1.f + __expf(-y.z));
        y.w = y.w / (1.f + __expf(-y.w));
        stg_v4f(Y + base + i * 4, y);
    }
}

// GroupNorm2 fused with FiLM conditioning and residual add; writes h.
__global__ __launch_bounds__(256) void gn2_fuse_kernel(
    const float* __restrict__ X, const float* __restrict__ gamma,
    const float* __restrict__ beta, const float* __restrict__ cond,
    const float* __restrict__ resid, float* __restrict__ H)
{
    const int bg = blockIdx.x;
    const int b = bg >> 6, grp = bg & 63;
    const size_t base = ((size_t)b * C_ + grp * 8) * N_;
    const int tid = threadIdx.x;
    v4f xc[8];
    float s1 = 0.f, s2 = 0.f;
    #pragma unroll
    for (int e = 0; e < 8; ++e) {
        int i = tid + e * 256;
        v4f v = ldg_v4f(X + base + i * 4);
        xc[e] = v;
        s1 += v.x + v.y + v.z + v.w;
        s2 += v.x * v.x + v.y * v.y + v.z * v.z + v.w * v.w;
    }
    float mean, rstd;
    gn_stats(s1, s2, mean, rstd);
    #pragma unroll
    for (int e = 0; e < 8; ++e) {
        int i = tid + e * 256;
        int c = grp * 8 + (i >> 8);
        float gm = ldg_f(&gamma[c]), bt = ldg_f(&beta[c]);
        float film = 1.f + ldg_f(&cond[b * C_ + c]);
        float ga = gm * rstd * film;
        float be = (bt - mean * rstd * gm) * film;
        v4f v = xc[e];
        v4f r = ldg_v4f(resid + base + i * 4), y;
        y.x = v.x * ga + be + r.x;
        y.y = v.y * ga + be + r.y;
        y.z = v.z * ga + be + r.z;
        y.w = v.w * ga + be + r.w;
        stg_v4f(H + base + i * 4, y);
    }
}

// ---------------------------------------------------------------------------
// Flash attention per (batch, head).  64-query tile per block, streaming over
// 16 key tiles of 64 with online softmax.  S = Q^T K (WMMA), O += V P^T (WMMA).
// K/V staging is software-pipelined: loads for tile m+1 fly during tile m's
// WMMAs + softmax; LDS scatter offsets and pointers hoisted out of the loop.
// ---------------------------------------------------------------------------
__global__ __launch_bounds__(256) void attn_flash(
    const float* __restrict__ Q, const float* __restrict__ Km,
    const float* __restrict__ V, float* __restrict__ O)
{
    const int n0 = blockIdx.x * 64;
    const int bh = blockIdx.y;                 // b*HEADS + h
    const int b = bh >> 3, hh = bh & 7;
    const size_t base = ((size_t)b * C_ + hh * D_) * N_;
    Q += base; Km += base; V += base; O += base;

    __shared__ alignas(16) _Float16 Qs[64 * 72];  // [nq][d]
    __shared__ alignas(16) _Float16 Ks[64 * 72];  // [m][d]   (B-frag layout)
    __shared__ alignas(16) _Float16 Vs[64 * 72];  // [d][m]   (A-frag layout)
    __shared__ alignas(16) _Float16 Ps[64 * 72];  // [nq][m]  (B-frag layout)
    __shared__ float Sld[64 * 65];
    __shared__ float row_max[64], row_sum[64], corr[64];
    __shared__ float pmax[64 * 4], psum[64 * 4];

    const int t = threadIdx.x, wave = t >> 5, lane = t & 31;
    const int g = lane >> 4, lr = lane & 15;
    const int mi  = wave & 3;            // tile row (nq for S, d for O)
    const int nib = (wave >> 2) * 2;     // tile col base (m for S, nq for O)

    v8f oacc[2] = {};

    // ---- hoisted K/V staging pointers + LDS offsets ----
    const float* pK[4];
    const float* pV[4];
    int kOfs[4], vOfs[4];
    #pragma unroll
    for (int e = 0; e < 4; ++e) {
        int lin4 = t + e * 256;               // 1024 float4 slots (64x64)
        int d   = lin4 >> 4;                  // 16 float4 per d-row
        int m4  = (lin4 & 15) * 4;
        pK[e] = Km + (size_t)d * N_ + m4;
        pV[e] = V + (size_t)d * N_ + m4;
        kOfs[e] = m4 * 72 + d;                // scatter (transpose)
        vOfs[e] = d * 72 + m4;                // packed
    }
    v4f kReg[4], vReg[4];
    auto loadKV = [&]() {
        #pragma unroll
        for (int e = 0; e < 4; ++e) {
            kReg[e] = ldg_v4f(pK[e]); pK[e] += 64;
            vReg[e] = ldg_v4f(pV[e]); pV[e] += 64;
        }
    };
    auto storeKV = [&]() {
        #pragma unroll
        for (int e = 0; e < 4; ++e) {
            v4h kh = __builtin_convertvector(kReg[e], v4h);
            Ks[kOfs[e]]           = kh.x;
            Ks[kOfs[e] + 72]      = kh.y;
            Ks[kOfs[e] + 144]     = kh.z;
            Ks[kOfs[e] + 216]     = kh.w;
            *(v4h*)&Vs[vOfs[e]] = __builtin_convertvector(vReg[e], v4h);
        }
    };

    // Q tile: transpose via float4 load + 4 scatter b16 stores
    #pragma unroll
    for (int e = 0; e < 4; ++e) {
        int lin4 = t + e * 256;
        int d   = lin4 >> 4;
        int nq4 = (lin4 & 15) * 4;
        v4f v = ldg_v4f(&Q[(size_t)d * N_ + n0 + nq4]);
        v4h h = __builtin_convertvector(v, v4h);
        Qs[(nq4 + 0) * 72 + d] = h.x;
        Qs[(nq4 + 1) * 72 + d] = h.y;
        Qs[(nq4 + 2) * 72 + d] = h.z;
        Qs[(nq4 + 3) * 72 + d] = h.w;
    }
    if (t < 64) { row_max[t] = -1e30f; row_sum[t] = 0.f; }
    loadKV();
    __syncthreads();

    const float scale = 0.125f;  // d^-0.5, d = 64

    for (int m0 = 0; m0 < N_; m0 += 64) {
        storeKV();                       // waits loadcnt, fills Ks/Vs
        __syncthreads();
        if (m0 + 64 < N_) loadKV();      // next tile's loads fly during compute

        // ---- S = Q^T K : M=nq, N=m, K=d (2 steps of 32) ----
        v8f sacc[2] = {};
        #pragma unroll
        for (int ks = 0; ks < 2; ++ks) {
            Frag16 a;
            const _Float16* ap = &Qs[(mi * 16 + lr) * 72 + ks * 32 + g * 8];
            a.h[0] = *(const v8h*)ap; a.h[1] = *(const v8h*)(ap + 16);
            #pragma unroll
            for (int ni = 0; ni < 2; ++ni) {
                Frag16 bf;
                const _Float16* bp = &Ks[((nib + ni) * 16 + lr) * 72 + ks * 32 + g * 8];
                bf.h[0] = *(const v8h*)bp; bf.h[1] = *(const v8h*)(bp + 16);
                sacc[ni] = wmma_f16f32(a, bf, sacc[ni]);
            }
        }
        #pragma unroll
        for (int ni = 0; ni < 2; ++ni)
            #pragma unroll
            for (int r = 0; r < 8; ++r)
                Sld[(mi * 16 + r + 8 * g) * 65 + (nib + ni) * 16 + lr] =
                    sacc[ni][r] * scale;
        __syncthreads();

        // ---- online softmax over this 64-wide key tile ----
        const int srow = t >> 2, part = t & 3;
        float mx = -1e30f;
        for (int j = part * 16; j < part * 16 + 16; ++j)
            mx = fmaxf(mx, Sld[srow * 65 + j]);
        pmax[srow * 4 + part] = mx;
        __syncthreads();
        if (part == 0) {
            float tm = fmaxf(fmaxf(pmax[srow*4], pmax[srow*4+1]),
                             fmaxf(pmax[srow*4+2], pmax[srow*4+3]));
            float nm = fmaxf(row_max[srow], tm);
            corr[srow] = __expf(row_max[srow] - nm);
            row_max[srow] = nm;
        }
        __syncthreads();
        float ps = 0.f;
        const float nm = row_max[srow];
        for (int j = part * 16; j < part * 16 + 16; ++j) {
            float p = __expf(Sld[srow * 65 + j] - nm);
            Ps[srow * 72 + j] = (_Float16)p;
            ps += p;
        }
        psum[srow * 4 + part] = ps;
        __syncthreads();
        if (part == 0)
            row_sum[srow] = row_sum[srow] * corr[srow] +
                (psum[srow*4] + psum[srow*4+1] + psum[srow*4+2] + psum[srow*4+3]);
        __syncthreads();

        // ---- O = O*corr + V P^T : M=d, N=nq, K=m (2 steps of 32) ----
        #pragma unroll
        for (int ni = 0; ni < 2; ++ni) {
            float c = corr[(nib + ni) * 16 + lr];
            #pragma unroll
            for (int r = 0; r < 8; ++r) oacc[ni][r] *= c;
        }
        #pragma unroll
        for (int ks = 0; ks < 2; ++ks) {
            Frag16 a;
            const _Float16* ap = &Vs[(mi * 16 + lr) * 72 + ks * 32 + g * 8];
            a.h[0] = *(const v8h*)ap; a.h[1] = *(const v8h*)(ap + 16);
            #pragma unroll
            for (int ni = 0; ni < 2; ++ni) {
                Frag16 bf;
                const _Float16* bp = &Ps[((nib + ni) * 16 + lr) * 72 + ks * 32 + g * 8];
                bf.h[0] = *(const v8h*)bp; bf.h[1] = *(const v8h*)(bp + 16);
                oacc[ni] = wmma_f16f32(a, bf, oacc[ni]);
            }
        }
        __syncthreads();
    }

    // ---- finalize ----
    #pragma unroll
    for (int ni = 0; ni < 2; ++ni) {
        float inv = 1.0f / row_sum[(nib + ni) * 16 + lr];
        #pragma unroll
        for (int r = 0; r < 8; ++r) {
            int d = mi * 16 + r + 8 * g;
            int n = n0 + (nib + ni) * 16 + lr;
            stg_f(&O[(size_t)d * N_ + n], oacc[ni][r] * inv);
        }
    }
}

// ---------------------------------------------------------------------------
extern "C" void kernel_launch(void* const* d_in, const int* in_sizes, int n_in,
                              void* d_out, int out_size, void* d_ws, size_t ws_size,
                              hipStream_t stream) {
    (void)in_sizes; (void)n_in; (void)out_size; (void)ws_size;
    const float* x    = (const float*)d_in[0];
    const float* cnd  = (const float*)d_in[1];
    const float* c1w  = (const float*)d_in[2];
    const float* c1b  = (const float*)d_in[3];
    const float* g1w  = (const float*)d_in[4];
    const float* g1b  = (const float*)d_in[5];
    const float* c2w  = (const float*)d_in[6];
    const float* c2b  = (const float*)d_in[7];
    const float* g2w  = (const float*)d_in[8];
    const float* g2b  = (const float*)d_in[9];
    const float* cpw  = (const float*)d_in[10];
    const float* cpb  = (const float*)d_in[11];
    const float* rpw  = (const float*)d_in[12];
    const float* rpb  = (const float*)d_in[13];
    const float* qw   = (const float*)d_in[14];
    const float* qb   = (const float*)d_in[15];
    const float* kw   = (const float*)d_in[16];
    const float* kb   = (const float*)d_in[17];
    const float* vw   = (const float*)d_in[18];
    const float* vb   = (const float*)d_in[19];
    const float* ow   = (const float*)d_in[20];
    const float* ob   = (const float*)d_in[21];

    const size_t BUF = (size_t)B_ * C_ * N_;      // 4M floats = 16 MB
    float* resid = (float*)d_ws;
    float* buf1  = resid + BUF;
    float* buf2  = buf1 + BUF;
    float* buf3  = buf2 + BUF;
    float* buf4  = buf3 + BUF;
    float* buf5  = buf4 + BUF;
    float* condv = buf5 + BUF;                    // B_*C_ floats

    dim3 blk(256);
    dim3 ggrid(N_ / 128, C_ / 128, B_);           // (8, 4, 8)

    // cond[b][c]
    cond_proj_kernel<<<(B_ * C_) / 256, blk, 0, stream>>>(cnd, cpw, cpb, condv);
    // residual = 1x1 conv(x)
    gemm_wmma<0, false><<<ggrid, blk, 0, stream>>>(rpw, x, rpb, nullptr, resid, C_, C_);
    // h1 = conv3x3(x)
    gemm_wmma<1, false><<<ggrid, blk, 0, stream>>>(c1w, x, c1b, nullptr, buf1, C_, C_ * 9);
    // gn1 + silu
    gn_silu_kernel<<<B_ * GROUPS_, blk, 0, stream>>>(buf1, g1w, g1b, buf2);
    // h2 = conv3x3(h1s)
    gemm_wmma<1, false><<<ggrid, blk, 0, stream>>>(c2w, buf2, c2b, nullptr, buf1, C_, C_ * 9);
    // h = gn2(h2) * (1 + cond) + residual   (writes buf2)
    gn2_fuse_kernel<<<B_ * GROUPS_, blk, 0, stream>>>(buf1, g2w, g2b, condv, resid, buf2);
    // q, k, v projections
    gemm_wmma<0, false><<<ggrid, blk, 0, stream>>>(qw, buf2, qb, nullptr, buf3, C_, C_);
    gemm_wmma<0, false><<<ggrid, blk, 0, stream>>>(kw, buf2, kb, nullptr, buf4, C_, C_);
    gemm_wmma<0, false><<<ggrid, blk, 0, stream>>>(vw, buf2, vb, nullptr, buf5, C_, C_);
    // attention -> buf1
    attn_flash<<<dim3(N_ / 64, B_ * HEADS_), blk, 0, stream>>>(buf3, buf4, buf5, buf1);
    // out = 1x1 conv(attn) + h
    gemm_wmma<0, true><<<ggrid, blk, 0, stream>>>(ow, buf1, ob, buf2, (float*)d_out, C_, C_);
}
